// Block_71640054497469
// MI455X (gfx1250) — matrix operand order repaired and checked by reading, not compiled, not run on hardware
//
#include <hip/hip_runtime.h>
#include <cstdint>
#include <cstddef>

typedef unsigned short u16;
typedef __attribute__((ext_vector_type(16))) __bf16 v16bf;
typedef __attribute__((ext_vector_type(8)))  float  v8f;

#define DIM 1024
#define HEADS 16
#define HEAD_DIM 64
#define HIDDEN 4096
#define BATCH 16
#define SEQ 577
#define TOK (BATCH * SEQ)   /* 9232, divisible by 16 */
#define NPAD 608            /* 19*32: padded key length for P*V chunks */
#define QT 37               /* ceil(577/16) */
#define LN_EPS 1e-5f

union Frag {
  v16bf v;
  u16   u[16];
  uint4 q[2];
};

__device__ __forceinline__ u16 f2bf(float f) {
  unsigned u = __float_as_uint(f);
  u += 0x7FFFu + ((u >> 16) & 1u);   // round to nearest even
  return (u16)(u >> 16);
}

__device__ __forceinline__ v8f v8zero() {
  v8f z = {0.f, 0.f, 0.f, 0.f, 0.f, 0.f, 0.f, 0.f};
  return z;
}

__device__ __forceinline__ v8f wmma_bf16(v16bf a, v16bf b, v8f c) {
  // (neg_a, A, neg_b, B, c_mod, C, reuse_a, reuse_b)
  return __builtin_amdgcn_wmma_f32_16x16x32_bf16(false, a, false, b,
                                                 (short)0, c, false, false);
}

// ---- CDNA5 async global->LDS copy (ASYNCcnt-tracked, bypasses VGPRs) ----
// LDS flat addresses map to LDS offsets via low 32 bits (aperture rule).
__device__ __forceinline__ void async_b128(unsigned lds_addr, const void* gptr) {
  unsigned long long ga = (unsigned long long)(uintptr_t)gptr;
  asm volatile("global_load_async_to_lds_b128 %0, %1, off"
               :
               : "v"(lds_addr), "v"(ga)
               : "memory");
}

__device__ __forceinline__ void wait_async() {
  asm volatile("s_wait_asynccnt 0x0" ::: "memory");
}

// ---------------------------------------------------------------- fp32->bf16
__global__ __launch_bounds__(256) void cvt_f32_bf16(const float* __restrict__ in,
                                                    u16* __restrict__ out, int n) {
  int i = (blockIdx.x * 256 + threadIdx.x) * 4;
  if (i + 3 < n) {
    float4 v = *(const float4*)(in + i);
    out[i + 0] = f2bf(v.x);
    out[i + 1] = f2bf(v.y);
    out[i + 2] = f2bf(v.z);
    out[i + 3] = f2bf(v.w);
  }
}

// ---------------------------------------------------------------- layernorm
__global__ __launch_bounds__(256) void layernorm_bf16(const float* __restrict__ x,
                                                      const float* __restrict__ g,
                                                      const float* __restrict__ bta,
                                                      u16* __restrict__ out) {
  __shared__ float redS[8], redQ[8], stats[2];
  int row = blockIdx.x;
  int t = threadIdx.x;
  const float* xr = x + (size_t)row * DIM;
  float4 v = *(const float4*)(xr + t * 4);
  float s = v.x + v.y + v.z + v.w;
  float q = v.x * v.x + v.y * v.y + v.z * v.z + v.w * v.w;
#pragma unroll
  for (int off = 16; off >= 1; off >>= 1) {
    s += __shfl_xor(s, off, 32);
    q += __shfl_xor(q, off, 32);
  }
  if ((t & 31) == 0) { redS[t >> 5] = s; redQ[t >> 5] = q; }
  __syncthreads();
  if (t == 0) {
    float ts = 0.f, tq = 0.f;
#pragma unroll
    for (int i = 0; i < 8; i++) { ts += redS[i]; tq += redQ[i]; }
    float mean = ts * (1.0f / DIM);
    float var = tq * (1.0f / DIM) - mean * mean;
    stats[0] = mean;
    stats[1] = rsqrtf(var + LN_EPS);
  }
  __syncthreads();
  float mean = stats[0], rstd = stats[1];
  int c = t * 4;
  float4 gg = *(const float4*)(g + c);
  float4 bb = *(const float4*)(bta + c);
  u16* orow = out + (size_t)row * DIM + c;
  orow[0] = f2bf((v.x - mean) * rstd * gg.x + bb.x);
  orow[1] = f2bf((v.y - mean) * rstd * gg.y + bb.y);
  orow[2] = f2bf((v.z - mean) * rstd * gg.z + bb.z);
  orow[3] = f2bf((v.w - mean) * rstd * gg.w + bb.w);
}

// ---------------------------------------------------------------- WMMA GEMM
// out = A[M,K] * W[N,K]^T + bias; K-step 64, double-buffered async staging.
#define EPI_BF16 0
#define EPI_RES_F32 1
#define EPI_GELU_BF16 2

template <int EPI>
__global__ __launch_bounds__(256) void gemm_wmma(const u16* __restrict__ A,
                                                 const u16* __restrict__ W,
                                                 const float* __restrict__ bias,
                                                 const float* __restrict__ res,
                                                 void* __restrict__ outp,
                                                 int M, int K, int Nout) {
  __shared__ __align__(16) u16 As[2][64 * 64];    // 2 x 8 KB
  __shared__ __align__(16) u16 Ws[2][128 * 64];   // 2 x 16 KB
  int t = threadIdx.x;
  int lane = t & 31, w = t >> 5;
  int wm = w >> 2, wn = w & 3;                    // 2 x 4 wave grid
  int nBlk = Nout >> 7;
  int mb = (blockIdx.x / nBlk) * 64;
  int nb = (blockIdx.x % nBlk) * 128;

  v8f acc[2][2];
  acc[0][0] = v8zero(); acc[0][1] = v8zero();
  acc[1][0] = v8zero(); acc[1][1] = v8zero();

  int mi = lane & 15;
  int kbA = (lane & 16) ? 8 : 0;
  int kbB = (lane & 16) ? 16 : 0;

  // per-thread staging coordinates (clamped rows: OOB results are never stored)
  int aRow[2], aCol[2];
#pragma unroll
  for (int j = 0; j < 2; j++) {
    int idx = t * 2 + j;
    aRow[j] = idx >> 3;
    aCol[j] = (idx & 7) * 8;
  }
  int wRow[4], wCol[4];
#pragma unroll
  for (int j = 0; j < 4; j++) {
    int idx = t * 4 + j;
    wRow[j] = idx >> 3;
    wCol[j] = (idx & 7) * 8;
  }

  auto stage = [&](int bufi, int k0) {
#pragma unroll
    for (int j = 0; j < 2; j++) {
      int gr = mb + aRow[j];
      if (gr >= M) gr = M - 1;
      async_b128((unsigned)(uintptr_t)(&As[bufi][aRow[j] * 64 + aCol[j]]),
                 A + (size_t)gr * K + k0 + aCol[j]);
    }
#pragma unroll
    for (int j = 0; j < 4; j++) {
      async_b128((unsigned)(uintptr_t)(&Ws[bufi][wRow[j] * 64 + wCol[j]]),
                 W + (size_t)(nb + wRow[j]) * K + k0 + wCol[j]);
    }
  };

  stage(0, 0);
  wait_async();
  __syncthreads();

  int nIter = K >> 6;
  for (int it = 0; it < nIter; it++) {
    int cur = it & 1;
    if (it + 1 < nIter) stage(cur ^ 1, (it + 1) << 6);  // overlap copy w/ math
    const u16* ab = &As[cur][0];
    const u16* wb = &Ws[cur][0];
#pragma unroll
    for (int ks = 0; ks < 2; ks++) {
      Frag af[2], bfr[2];
#pragma unroll
      for (int s2 = 0; s2 < 2; s2++) {
        const u16* ar = ab + (wm * 32 + s2 * 16 + mi) * 64 + ks * 32;
        af[s2].q[0] = *(const uint4*)(ar + kbA);        // K = kbA..kbA+7
        af[s2].q[1] = *(const uint4*)(ar + 16 + kbA);   // K = 16+kbA..
        const u16* wr = wb + (wn * 32 + s2 * 16 + mi) * 64 + ks * 32;
        bfr[s2].q[0] = *(const uint4*)(wr + kbA);       // K = kbA..kbA+7
        bfr[s2].q[1] = *(const uint4*)(wr + 16 + kbA);
      }
#pragma unroll
      for (int i = 0; i < 2; i++)
#pragma unroll
        for (int j = 0; j < 2; j++)
          acc[i][j] = wmma_bf16(af[i].v, bfr[j].v, acc[i][j]);
    }
    wait_async();
    __syncthreads();
  }

  // epilogue: C/D layout -> lanes 0-15: M=r, lanes 16-31: M=r+8, N=lane%16
#pragma unroll
  for (int i = 0; i < 2; i++) {
#pragma unroll
    for (int j = 0; j < 2; j++) {
      int mbase = mb + wm * 32 + i * 16 + ((lane & 16) ? 8 : 0);
      int n = nb + wn * 32 + j * 16 + (lane & 15);
      float bv = bias[n];
#pragma unroll
      for (int r = 0; r < 8; r++) {
        int mm = mbase + r;
        if (mm < M) {
          float vv = acc[i][j][r] + bv;
          size_t idx = (size_t)mm * Nout + n;
          if (EPI == EPI_GELU_BF16) {
            vv = 0.5f * vv * (1.0f + erff(vv * 0.70710678f));
            ((u16*)outp)[idx] = f2bf(vv);
          } else if (EPI == EPI_BF16) {
            ((u16*)outp)[idx] = f2bf(vv);
          } else {
            ((float*)outp)[idx] = vv + res[idx];
          }
        }
      }
    }
  }
}

// ---------------------------------------------------------------- attention
// one wave per (batch, head, 16-query tile); qkv is bf16 [TOK][3*DIM]
__global__ __launch_bounds__(32) void attn_wmma(const u16* __restrict__ qkv,
                                                u16* __restrict__ y) {
  __shared__ __align__(16) float sc[16 * NPAD];  // 38912 B score/prob buffer
  __shared__ __align__(16) u16 kv[32 * 64];      // 4096 B K/V staging
  int lane = threadIdx.x;
  int qt = blockIdx.x % QT;
  int h = (blockIdx.x / QT) % HEADS;
  int b = blockIdx.x / (QT * HEADS);
  int mi = lane & 15;
  int hi = lane >> 4;  // 0/1
  const size_t rstr = 3 * DIM;
  const uint4 z4 = {0u, 0u, 0u, 0u};

  // ---- Q fragments (A layout), reused for every key tile
  int qtok = qt * 16 + mi;
  const u16* qrow =
      qkv + ((size_t)(b * SEQ + (qtok < SEQ ? qtok : 0))) * rstr + h * HEAD_DIM;
  int kbA = hi * 8;
  Frag a0, a1;
  a0.q[0] = *(const uint4*)(qrow + kbA);
  a0.q[1] = *(const uint4*)(qrow + 16 + kbA);
  a1.q[0] = *(const uint4*)(qrow + 32 + kbA);
  a1.q[1] = *(const uint4*)(qrow + 48 + kbA);
  if (qtok >= SEQ) { a0.q[0] = z4; a0.q[1] = z4; a1.q[0] = z4; a1.q[1] = z4; }

  // mask pad columns 592..607
  if (lane < 16) {
#pragma unroll
    for (int r = 0; r < 16; r++) sc[r * NPAD + 592 + lane] = -1e30f;
  }

  // ---- scores = Q * K^T * scale
  const float scale = 0.125f;  // 1/sqrt(64)
  int kbB = hi * 16;
  for (int kt = 0; kt < QT; kt++) {
    int key = kt * 16 + mi;
    int keyc = key < SEQ ? key : SEQ - 1;  // clamp; invalid cols masked below
    const u16* ksrc = qkv + ((size_t)(b * SEQ + keyc)) * rstr +
                      DIM + h * HEAD_DIM + hi * 32;
    unsigned kl = (unsigned)(uintptr_t)(kv + mi * 64 + hi * 32);
#pragma unroll
    for (int j2 = 0; j2 < 4; j2++)
      async_b128(kl + j2 * 16, ksrc + j2 * 8);
    wait_async();
    __syncthreads();

    Frag b0, b1;
    const u16* krow = kv + mi * 64;  // row = key-in-tile, contiguous depth
    b0.q[0] = *(const uint4*)(krow + kbA);
    b0.q[1] = *(const uint4*)(krow + 16 + kbA);
    b1.q[0] = *(const uint4*)(krow + 32 + kbA);
    b1.q[1] = *(const uint4*)(krow + 48 + kbA);

    v8f c = v8zero();
    c = wmma_bf16(a0.v, b0.v, c);
    c = wmma_bf16(a1.v, b1.v, c);
    __syncthreads();

    int kidx = kt * 16 + mi;
    bool kval = kidx < SEQ;
    int mrow0 = hi * 8;
#pragma unroll
    for (int r = 0; r < 8; r++)
      sc[(mrow0 + r) * NPAD + kidx] = kval ? c[r] * scale : -1e30f;
  }
  __syncthreads();

  // ---- softmax: row = mi, lane pair (hi) splits columns
  float* srow = sc + mi * NPAD;
  float mx = -1e30f;
  for (int j2 = hi; j2 < NPAD / 4; j2 += 2) {
    float4 v = *(const float4*)(srow + j2 * 4);
    mx = fmaxf(mx, fmaxf(fmaxf(v.x, v.y), fmaxf(v.z, v.w)));
  }
  mx = fmaxf(mx, __shfl_xor(mx, 16, 32));
  float sum = 0.f;
  for (int j2 = hi; j2 < NPAD / 4; j2 += 2) {
    float4 v = *(const float4*)(srow + j2 * 4);
    v.x = __expf(v.x - mx); v.y = __expf(v.y - mx);
    v.z = __expf(v.z - mx); v.w = __expf(v.w - mx);
    *(float4*)(srow + j2 * 4) = v;
    sum += v.x + v.y + v.z + v.w;
  }
  sum += __shfl_xor(sum, 16, 32);
  float inv = 1.0f / sum;
  for (int j2 = hi; j2 < NPAD / 4; j2 += 2) {
    float4 v = *(const float4*)(srow + j2 * 4);
    v.x *= inv; v.y *= inv; v.z *= inv; v.w *= inv;
    *(float4*)(srow + j2 * 4) = v;
  }
  __syncthreads();

  // ---- y = P * V  (19 chunks of 32 keys, 4 column tiles of 16)
  // invalid key rows have probability exactly 0, so clamped V rows are harmless
  v8f accv[4];
  accv[0] = v8zero(); accv[1] = v8zero(); accv[2] = v8zero(); accv[3] = v8zero();
  for (int ck = 0; ck < NPAD / 32; ck++) {
    int vrow = ck * 32 + lane;
    int vrc = vrow < SEQ ? vrow : SEQ - 1;
    const u16* vsrc = qkv + ((size_t)(b * SEQ + vrc)) * rstr +
                      2 * DIM + h * HEAD_DIM;
    unsigned vl = (unsigned)(uintptr_t)(kv + lane * 64);
#pragma unroll
    for (int j2 = 0; j2 < 8; j2++)
      async_b128(vl + j2 * 16, vsrc + j2 * 8);
    wait_async();
    __syncthreads();

    Frag pa;
    const float* pbase = sc + mi * NPAD + ck * 32;
#pragma unroll
    for (int t2 = 0; t2 < 8; t2++) {
      pa.u[t2] = f2bf(pbase[kbA + t2]);           // K = kbA..kbA+7
      pa.u[8 + t2] = f2bf(pbase[16 + kbA + t2]);  // K = 16+kbA..
    }
#pragma unroll
    for (int nb2 = 0; nb2 < 4; nb2++) {
      Frag vb;
#pragma unroll
      for (int i2 = 0; i2 < 16; i2++)
        vb.u[i2] = kv[((i2 < 8) ? (kbA + i2) : (16 + kbA + i2 - 8)) * 64 + nb2 * 16 + mi];  // B[k][n] = V[k][n]
      accv[nb2] = wmma_bf16(pa.v, vb.v, accv[nb2]);
    }
    __syncthreads();
  }

  // ---- store y tile (bf16)
#pragma unroll
  for (int nb2 = 0; nb2 < 4; nb2++) {
#pragma unroll
    for (int r = 0; r < 8; r++) {
      int tok = qt * 16 + hi * 8 + r;
      if (tok < SEQ)
        y[((size_t)(b * SEQ + tok)) * DIM + h * HEAD_DIM + nb2 * 16 + mi] =
            f2bf(accv[nb2][r]);
    }
  }
}

// ---------------------------------------------------------------- launcher
extern "C" void kernel_launch(void* const* d_in, const int* in_sizes, int n_in,
                              void* d_out, int out_size, void* d_ws, size_t ws_size,
                              hipStream_t stream) {
  (void)in_sizes; (void)n_in; (void)out_size; (void)ws_size;
  const float* x      = (const float*)d_in[0];
  const float* w_qkv  = (const float*)d_in[1];
  const float* b_qkv  = (const float*)d_in[2];
  const float* w_proj = (const float*)d_in[3];
  const float* b_proj = (const float*)d_in[4];
  const float* ln1_g  = (const float*)d_in[5];
  const float* ln1_b  = (const float*)d_in[6];
  const float* ln2_g  = (const float*)d_in[7];
  const float* ln2_b  = (const float*)d_in[8];
  const float* w_fc1  = (const float*)d_in[9];
  const float* b_fc1  = (const float*)d_in[10];
  const float* w_fc2  = (const float*)d_in[11];
  const float* b_fc2  = (const float*)d_in[12];
  float* out = (float*)d_out;

  char* ws = (char*)d_ws;
  size_t off = 0;
  auto take = [&](size_t bytes) -> char* {
    char* p = ws + off;
    off += (bytes + 255) & ~(size_t)255;
    return p;
  };
  u16*   xn     = (u16*)take((size_t)TOK * DIM * 2);
  u16*   qkvb   = (u16*)take((size_t)TOK * 3 * DIM * 2);
  u16*   yb     = (u16*)take((size_t)TOK * DIM * 2);
  float* x1     = (float*)take((size_t)TOK * DIM * 4);
  u16*   xn2    = (u16*)take((size_t)TOK * DIM * 2);
  u16*   hb     = (u16*)take((size_t)TOK * HIDDEN * 2);
  u16*   wqkvb  = (u16*)take((size_t)3 * DIM * DIM * 2);
  u16*   wprojb = (u16*)take((size_t)DIM * DIM * 2);
  u16*   wfc1b  = (u16*)take((size_t)HIDDEN * DIM * 2);
  u16*   wfc2b  = (u16*)take((size_t)DIM * HIDDEN * 2);

  cvt_f32_bf16<<<3 * DIM * DIM / 1024, 256, 0, stream>>>(w_qkv, wqkvb, 3 * DIM * DIM);
  cvt_f32_bf16<<<DIM * DIM / 1024, 256, 0, stream>>>(w_proj, wprojb, DIM * DIM);
  cvt_f32_bf16<<<HIDDEN * DIM / 1024, 256, 0, stream>>>(w_fc1, wfc1b, HIDDEN * DIM);
  cvt_f32_bf16<<<DIM * HIDDEN / 1024, 256, 0, stream>>>(w_fc2, wfc2b, DIM * HIDDEN);

  layernorm_bf16<<<TOK, 256, 0, stream>>>(x, ln1_g, ln1_b, xn);

  int mTiles = (TOK + 63) / 64;  // 145
  gemm_wmma<EPI_BF16><<<mTiles * (3 * DIM / 128), 256, 0, stream>>>(
      xn, wqkvb, b_qkv, nullptr, qkvb, TOK, DIM, 3 * DIM);

  attn_wmma<<<BATCH * HEADS * QT, 32, 0, stream>>>(qkvb, yb);

  gemm_wmma<EPI_RES_F32><<<mTiles * (DIM / 128), 256, 0, stream>>>(
      yb, wprojb, b_proj, x, x1, TOK, DIM, DIM);

  layernorm_bf16<<<TOK, 256, 0, stream>>>(x1, ln2_g, ln2_b, xn2);

  gemm_wmma<EPI_GELU_BF16><<<mTiles * (HIDDEN / 128), 256, 0, stream>>>(
      xn2, wfc1b, b_fc1, nullptr, hb, TOK, DIM, HIDDEN);

  gemm_wmma<EPI_RES_F32><<<mTiles * (DIM / 128), 256, 0, stream>>>(
      hb, wfc2b, b_fc2, x1, out, TOK, HIDDEN, DIM);
}
